// GlobalAttention_18176301597135
// MI455X (gfx1250) — matrix operand order repaired and checked by reading, not compiled, not run on hardware
//
#include <hip/hip_runtime.h>
#include <hip/hip_bf16.h>

typedef __attribute__((ext_vector_type(16))) _Float16 v16h;
typedef __attribute__((ext_vector_type(8)))  _Float16 v8h_t;
typedef __attribute__((ext_vector_type(4)))  _Float16 h4;
typedef __attribute__((ext_vector_type(8)))  float    v8f;
typedef __attribute__((ext_vector_type(4)))  float    v4f;

#define WMMA_F16(A,B,C) __builtin_amdgcn_wmma_f32_16x16x32_f16(false,(A),false,(B),(short)0,(C),false,false)

// Problem dims
#define BB 256
#define NN 2048
#define AA 256
#define HH 8
#define OO 256

// Workspace byte offsets
#define WS_GATEW   0u            // 256x256 f16 B-fragments (131072 B)
#define WS_OUTW    131072u       // 256x256 f16 B-fragments (131072 B)
#define WS_KEYW    262144u       // 256x32  f16 B-fragments (16384 B)
#define WS_VALW    278528u       // 256x32  f16 B-fragments (16384 B)
#define WS_QPART   294912u       // 256*16*256 f32 partial sums (4194304 B)
#define WS_MSUM    4489216u      // 256*16 f32 (16384 B)
#define WS_QVEC    4505600u      // 256*256 f32 q (262144 B)
#define WS_WAVG    4767744u      // 256*256 f32 weighted_avg (262144 B)

// ---------------------------------------------------------------------------
// A-operand fragment load from LDS (f16, padded row stride, 16x32 tile).
// ISA 16-bit A layout: lane L (row = rowBase + L%16) holds
//   elems 0..7  -> K = kBase + off + 0..7
//   elems 8..15 -> K = kBase + 16 + off + 0..7,  off = 8*(L>=16)
// ---------------------------------------------------------------------------
__device__ __forceinline__ v16h load_a_frag(const _Float16* lds, int rowBase,
                                            int rowStride, int kBase, int lane) {
  int row = rowBase + (lane & 15);
  int off = (lane >> 4) << 3;
  const _Float16* p = lds + row * rowStride + kBase + off;
  v8h_t lo = *(const v8h_t*)p;
  v8h_t hi = *(const v8h_t*)(p + 16);
  v16h a;
#pragma unroll
  for (int i = 0; i < 8; ++i) { a[i] = lo[i]; a[8 + i] = hi[i]; }
  return a;
}

// B-operand fragment: pre-packed, lane L reads 16 contiguous halves (32 B).
__device__ __forceinline__ v16h load_b_frag(const _Float16* g, int frag, int lane) {
  return ((const v16h*)g)[frag * 32 + lane];
}

__device__ __forceinline__ float sigm(float x) {
  return 1.0f / (1.0f + __expf(-x));
}

// ---------------------------------------------------------------------------
// Kernel 0: pack f32 row-major weights [K,N] into f16 WMMA B-fragments.
// frag = nt*(K/32)+kt ; within frag, half idx = L*16+i maps to
//   k = kt*32 + 8*(L>=16) + (i&7) + 16*(i>=8) ,  n = nt*16 + (L%16)
// ---------------------------------------------------------------------------
__device__ __forceinline__ void conv_one(_Float16* dst, const float* src,
                                         int K, int N, int idx) {
  int frag = idx >> 9;
  int rem  = idx & 511;
  int L = rem >> 4, i = rem & 15;
  int kfr = K >> 5;
  int nt = frag / kfr, kt = frag - nt * kfr;
  int k = (kt << 5) + ((L >> 4) << 3) + (i & 7) + ((i >> 3) << 4);
  int n = (nt << 4) + (L & 15);
  dst[idx] = (_Float16)src[k * N + n];
}

__global__ void __launch_bounds__(256) ga_pack_weights(
    const float* __restrict__ gating_w, const float* __restrict__ output_w,
    const float* __restrict__ key_w, const float* __restrict__ value_w,
    char* __restrict__ ws) {
  int id = blockIdx.x * 256 + threadIdx.x;
  if (id < 65536) {
    conv_one((_Float16*)(ws + WS_GATEW), gating_w, 256, 256, id);
  } else if (id < 131072) {
    conv_one((_Float16*)(ws + WS_OUTW), output_w, 256, 256, id - 65536);
  } else if (id < 139264) {
    conv_one((_Float16*)(ws + WS_KEYW), key_w, 256, 32, id - 131072);
  } else if (id < 147456) {
    conv_one((_Float16*)(ws + WS_VALW), value_w, 256, 32, id - 139264);
  }
}

// ---------------------------------------------------------------------------
// Kernel 1: masked column sums over N, float4-vectorized.
// ---------------------------------------------------------------------------
__global__ void __launch_bounds__(256) ga_qavg_partial(
    const float* __restrict__ q_data, const float* __restrict__ q_mask,
    char* __restrict__ ws) {
  float* qpart = (float*)(ws + WS_QPART);
  float* msum  = (float*)(ws + WS_MSUM);
  int bi = blockIdx.x;            // 0..1023
  int b = bi >> 2, ch = bi & 3;
  int tid = threadIdx.x;
  int sub = tid >> 6, c4 = tid & 63;
  const v4f* src = (const v4f*)(q_data + (size_t)b * NN * AA);
  v4f acc = {0.f, 0.f, 0.f, 0.f};
  float ms = 0.f;
  int n0 = ch * (NN / 4);
  for (int n = n0 + sub; n < n0 + NN / 4; n += 4) {
    float m = q_mask[(size_t)b * NN + n];
    v4f d = src[(size_t)n * 64 + c4];
    acc += d * m;
    ms += m;
  }
  *(v4f*)&qpart[((size_t)bi * 4 + sub) * 256 + (c4 << 2)] = acc;
  if (c4 == 0) msum[bi * 4 + sub] = ms;
}

// ---------------------------------------------------------------------------
// Kernel 2: combine 16 partials -> q_avg ; q = (q_avg @ query_w) * KD^-0.5.
// ---------------------------------------------------------------------------
__global__ void __launch_bounds__(256) ga_qproj(const float* __restrict__ query_w,
                                                char* __restrict__ ws) {
  const float* qpart = (const float*)(ws + WS_QPART);
  const float* msum  = (const float*)(ws + WS_MSUM);
  float* qvec = (float*)(ws + WS_QVEC);
  int b = blockIdx.x, tid = threadIdx.x;
  __shared__ float qa[AA];
  float s = 0.f, ms = 0.f;
#pragma unroll
  for (int c = 0; c < 16; ++c) {
    s  += qpart[((size_t)b * 16 + c) * 256 + tid];
    ms += msum[b * 16 + c];
  }
  qa[tid] = s / (ms + 1e-10f);
  __syncthreads();
  float acc = 0.f;
#pragma unroll 8
  for (int a = 0; a < AA; ++a) acc += qa[a] * query_w[a * 256 + tid];
  qvec[(size_t)b * 256 + tid] = acc * 0.17677669529663687f;  // 1/sqrt(32)
}

// ---------------------------------------------------------------------------
// Kernel 3: fused k/v projection + attention, one block per batch.
// B-fragments held in registers across the whole N loop; q read from LDS
// (uniform v4f loads) to keep VGPR pressure below the spill threshold.
// ---------------------------------------------------------------------------
__global__ void __launch_bounds__(256, 1) ga_attention(
    const float* __restrict__ m_data, const float* __restrict__ q_mask,
    char* __restrict__ ws) {
  const _Float16* keyF = (const _Float16*)(ws + WS_KEYW);
  const _Float16* valF = (const _Float16*)(ws + WS_VALW);
  const float* qvec = (const float*)(ws + WS_QVEC);
  float* wavg = (float*)(ws + WS_WAVG);

  __shared__ alignas(16) _Float16 m_tile[32 * 264];
  __shared__ alignas(16) float k_t[32][36];   // [j][kd]
  __shared__ alignas(16) float v_t[32][36];   // transposed: [vd][j]
  __shared__ alignas(16) float qh[HH][32];
  __shared__ float bias_l[NN];

  int b = blockIdx.x;
  int tid = threadIdx.x;
  int wave = tid >> 5, lane = tid & 31;

  qh[wave][lane] = qvec[(size_t)b * 256 + tid];
  for (int i = 0; i < NN / 256; ++i) {
    int idx = tid + i * 256;
    bias_l[idx] = 1e9f * (q_mask[(size_t)b * NN + idx] - 1.0f);
  }
  __syncthreads();

  int mat = wave >> 2;           // 0 = key, 1 = value
  int mt  = (wave >> 1) & 1;     // row tile
  int nt  = wave & 1;            // col tile
  const _Float16* wF = mat ? valF : keyF;

  // B-fragments are invariant over the whole N loop: keep in registers.
  v16h Bfr[8];
#pragma unroll
  for (int ks = 0; ks < 8; ++ks) Bfr[ks] = load_b_frag(wF, nt * 8 + ks, lane);

  float m_run = -1e30f, s_run = 0.f, acc = 0.f;
  const v4f* msrc = (const v4f*)(m_data + (size_t)b * NN * AA);
  const v4f* qrow = (const v4f*)&qh[wave][0];
  int colB = nt * 16 + (lane & 15);
  int rb = (lane >> 4) << 3;

  for (int t = 0; t < NN / 32; ++t) {
    int n0 = t * 32;
    // stage m_data tile (32x256) as f16: float4 load + packed b64 store
#pragma unroll
    for (int it = 0; it < 8; ++it) {
      int v = tid + it * 256;
      int row = v >> 6, c4 = v & 63;
      v4f d = msrc[(size_t)(n0 + row) * 64 + c4];
      h4 hd;
      hd[0] = (_Float16)d[0]; hd[1] = (_Float16)d[1];
      hd[2] = (_Float16)d[2]; hd[3] = (_Float16)d[3];
      *(h4*)&m_tile[row * 264 + (c4 << 2)] = hd;
    }
    if (t + 1 < NN / 32) {  // prefetch next tile into WGP caches
      const char* nxt = (const char*)(msrc + (size_t)(n0 + 32) * 64);
      __builtin_prefetch(nxt + tid * 128, 0, 3);
      __builtin_prefetch(nxt + tid * 128 + 64, 0, 3);
    }
    __syncthreads();

    // k/v tiles via WMMA; branchless C store (mat selects address only)
    {
      v8f c = {};
#pragma unroll
      for (int ks = 0; ks < 8; ++ks) {
        v16h Af = load_a_frag(m_tile, mt * 16, 264, ks * 32, lane);
        c = WMMA_F16(Af, Bfr[ks], c);
      }
#pragma unroll
      for (int r = 0; r < 8; ++r) {
        int row = mt * 16 + rb + r;
        float* dst = mat ? &v_t[colB][row] : &k_t[row][colB];
        *dst = c[r];
      }
    }
    __syncthreads();

    // online softmax: wave = head, lane = key position j in this tile
    {
      float logit = bias_l[n0 + lane];
      const v4f* krow = (const v4f*)&k_t[lane][0];
#pragma unroll
      for (int kq = 0; kq < 8; ++kq) {
        v4f kk = krow[kq];
        v4f qq = qrow[kq];
        logit += qq[0] * kk[0] + qq[1] * kk[1] + qq[2] * kk[2] + qq[3] * kk[3];
      }
      float tmax = logit;
#pragma unroll
      for (int o = 16; o > 0; o >>= 1) tmax = fmaxf(tmax, __shfl_xor(tmax, o, 32));
      float newm = fmaxf(m_run, tmax);
      float p = __expf(logit - newm);
      float corr = __expf(m_run - newm);
      float psum = p;
#pragma unroll
      for (int o = 16; o > 0; o >>= 1) psum += __shfl_xor(psum, o, 32);
      s_run = s_run * corr + psum;
      acc *= corr;
      const v4f* vrow = (const v4f*)&v_t[lane][0];  // lane = vd
#pragma unroll
      for (int jq = 0; jq < 8; ++jq) {
        v4f vv = vrow[jq];
        acc += __shfl(p, 4 * jq + 0, 32) * vv[0];
        acc += __shfl(p, 4 * jq + 1, 32) * vv[1];
        acc += __shfl(p, 4 * jq + 2, 32) * vv[2];
        acc += __shfl(p, 4 * jq + 3, 32) * vv[3];
      }
      m_run = newm;
    }
    __syncthreads();
  }
  wavg[(size_t)b * 256 + wave * 32 + lane] = acc / s_run;
}

// ---------------------------------------------------------------------------
// Kernel 4: fused gating + output projection, 32 rows of N per block.
// 2x2 register tile per wave: A fragments loaded once per k-step,
// reused for both N-tiles (nt = wave, wave+8).
// ---------------------------------------------------------------------------
__global__ void __launch_bounds__(256, 1) ga_gate_out(
    const float* __restrict__ q_data, const float* __restrict__ gating_b,
    const float* __restrict__ output_b, const char* __restrict__ ws,
    float* __restrict__ out) {
  const _Float16* gateF = (const _Float16*)(ws + WS_GATEW);
  const _Float16* outF  = (const _Float16*)(ws + WS_OUTW);
  const float* wavg = (const float*)(ws + WS_WAVG);

  __shared__ alignas(16) _Float16 q_tile[32 * 264];
  __shared__ alignas(16) _Float16 g_tile[32 * 264];

  int blk = blockIdx.x;
  int b = blk >> 6;
  int n0 = (blk & 63) * 32;
  int tid = threadIdx.x;
  int wave = tid >> 5, lane = tid & 31;

  // stage q_data tile (32x256) as f16: float4 load + packed b64 store
  const v4f* qsrc = (const v4f*)(q_data + ((size_t)b * NN + n0) * AA);
#pragma unroll
  for (int it = 0; it < 8; ++it) {
    int v = tid + it * 256;
    int row = v >> 6, c4 = v & 63;
    v4f d = qsrc[row * 64 + c4];
    h4 hd;
    hd[0] = (_Float16)d[0]; hd[1] = (_Float16)d[1];
    hd[2] = (_Float16)d[2]; hd[3] = (_Float16)d[3];
    *(h4*)&q_tile[row * 264 + (c4 << 2)] = hd;
  }
  __syncthreads();

  int ntA = wave, ntB = wave + 8;
  int rb = (lane >> 4) << 3;
  int lc = lane & 15;

  for (int sel = 0; sel < 2; ++sel) {
    const _Float16* wF = sel ? outF : gateF;
    const _Float16* aT = sel ? g_tile : q_tile;
    v8f a00 = {}, a10 = {}, a01 = {}, a11 = {};  // [mt][nt]
#pragma unroll
    for (int ks = 0; ks < 8; ++ks) {
      v16h A0 = load_a_frag(aT, 0, 264, ks * 32, lane);
      v16h A1 = load_a_frag(aT, 16, 264, ks * 32, lane);
      v16h B0 = load_b_frag(wF, ntA * 8 + ks, lane);
      v16h B1 = load_b_frag(wF, ntB * 8 + ks, lane);
      a00 = WMMA_F16(A0, B0, a00);
      a10 = WMMA_F16(A1, B0, a10);
      a01 = WMMA_F16(A0, B1, a01);
      a11 = WMMA_F16(A1, B1, a11);
    }
    if (sel == 0) {
      int colA = ntA * 16 + lc, colB = ntB * 16 + lc;
      float gbA = gating_b[colA], gbB = gating_b[colB];
      float waA = wavg[(size_t)b * 256 + colA], waB = wavg[(size_t)b * 256 + colB];
#pragma unroll
      for (int r = 0; r < 8; ++r) {
        g_tile[(rb + r) * 264 + colA]      = (_Float16)(sigm(a00[r] + gbA) * waA);
        g_tile[(16 + rb + r) * 264 + colA] = (_Float16)(sigm(a10[r] + gbA) * waA);
        g_tile[(rb + r) * 264 + colB]      = (_Float16)(sigm(a01[r] + gbB) * waB);
        g_tile[(16 + rb + r) * 264 + colB] = (_Float16)(sigm(a11[r] + gbB) * waB);
      }
    } else {
      int colA = ntA * 16 + lc, colB = ntB * 16 + lc;
      float obA = output_b[colA], obB = output_b[colB];
      float* orow = out + (size_t)(b * NN + n0) * OO;
#pragma unroll
      for (int r = 0; r < 8; ++r) {
        orow[(size_t)(rb + r) * OO + colA]      = a00[r] + obA;
        orow[(size_t)(16 + rb + r) * OO + colA] = a10[r] + obA;
        orow[(size_t)(rb + r) * OO + colB]      = a01[r] + obB;
        orow[(size_t)(16 + rb + r) * OO + colB] = a11[r] + obB;
      }
    }
    __syncthreads();
  }
}

// ---------------------------------------------------------------------------
extern "C" void kernel_launch(void* const* d_in, const int* in_sizes, int n_in,
                              void* d_out, int out_size, void* d_ws, size_t ws_size,
                              hipStream_t stream) {
  const float* q_data   = (const float*)d_in[0];
  const float* m_data   = (const float*)d_in[1];
  const float* q_mask   = (const float*)d_in[2];
  // d_in[3] (bias) is overwritten by the reference -> unused
  const float* query_w  = (const float*)d_in[4];
  const float* key_w    = (const float*)d_in[5];
  const float* value_w  = (const float*)d_in[6];
  const float* gating_w = (const float*)d_in[7];
  const float* gating_b = (const float*)d_in[8];
  const float* output_w = (const float*)d_in[9];
  const float* output_b = (const float*)d_in[10];
  float* out = (float*)d_out;
  char* ws = (char*)d_ws;

  ga_pack_weights<<<576, 256, 0, stream>>>(gating_w, output_w, key_w, value_w, ws);
  ga_qavg_partial<<<BB * 4, 256, 0, stream>>>(q_data, q_mask, ws);
  ga_qproj<<<BB, 256, 0, stream>>>(query_w, ws);
  ga_attention<<<BB, 256, 0, stream>>>(m_data, q_mask, ws);
  ga_gate_out<<<BB * (NN / 32), 256, 0, stream>>>(q_data, gating_b, output_b, ws, out);
}
